// FeatureAttention_34978213658711
// MI455X (gfx1250) — compile-verified, matrix operand order checked by reading, not run on hardware
//
#include <hip/hip_runtime.h>
#include <hip/hip_bf16.h>

// ---------------------------------------------------------------------------
// FeatureAttention on gfx1250 (MI455X, wave32, WMMA).
// Pipeline: [PROJ Q][PROJ K][PROJ V->V^T][SCORES][SOFTMAX][ATTN*V]
// GEMMs: 256x128x32 block tiles, 8 waves/WG, each wave = 64x64 (4x4 wmma
// tiles), v_wmma_f32_16x16x32_f16 with f32 accumulation.
//  - f32-input GEMMs (projections): register staging with f32->f16 convert.
//  - f16-input GEMMs (scores, attn*V): GLOBAL_LOAD_ASYNC_TO_LDS_B128 direct
//    global->LDS copies, synced with s_wait_asynccnt + barrier (ASYNCcnt).
// ---------------------------------------------------------------------------

typedef __attribute__((ext_vector_type(16))) _Float16 v16h;
typedef __attribute__((ext_vector_type(8)))  float    v8f;

#define BM   256         // block tile M
#define BN   128         // block tile N
#define BK   32          // K step
#define LDH  40          // LDS row stride in halves: 32 data + 8 pad (80B, 16B-aligned)

enum { MODE_PROJ = 0, MODE_PROJV = 1, MODE_SCORES = 2, MODE_AV = 3 };

static constexpr int Bsz = 4;
static constexpr int Ssz = 2048;
static constexpr int Dsz = 1024;
static constexpr int Usz = 1024;

struct StageRegs { int4 lo, hi; };   // 16 f16 values staged per thread-segment

// One 1x16 segment of a (ROWSx32) tile. seg: row = seg>>1, col = (seg&1)*16.
template<bool SRC_F32>
__device__ inline StageRegs load_stage_seg(const void* srcv, int ld, size_t rowOff,
                                           int k0, int seg) {
    const int row = seg >> 1;
    const int col = (seg & 1) * 16;
    StageRegs r;
    if constexpr (SRC_F32) {
        const float* p = (const float*)srcv + (rowOff + (size_t)row) * (size_t)ld + k0 + col;
        float4 f0 = *(const float4*)(p + 0);
        float4 f1 = *(const float4*)(p + 4);
        float4 f2 = *(const float4*)(p + 8);
        float4 f3 = *(const float4*)(p + 12);
        union { int4 q; _Float16 h[8]; } u0, u1;
        u0.h[0] = (_Float16)f0.x; u0.h[1] = (_Float16)f0.y;
        u0.h[2] = (_Float16)f0.z; u0.h[3] = (_Float16)f0.w;
        u0.h[4] = (_Float16)f1.x; u0.h[5] = (_Float16)f1.y;
        u0.h[6] = (_Float16)f1.z; u0.h[7] = (_Float16)f1.w;
        u1.h[0] = (_Float16)f2.x; u1.h[1] = (_Float16)f2.y;
        u1.h[2] = (_Float16)f2.z; u1.h[3] = (_Float16)f2.w;
        u1.h[4] = (_Float16)f3.x; u1.h[5] = (_Float16)f3.y;
        u1.h[6] = (_Float16)f3.z; u1.h[7] = (_Float16)f3.w;
        r.lo = u0.q; r.hi = u1.q;
    } else {
        const _Float16* p = (const _Float16*)srcv + (rowOff + (size_t)row) * (size_t)ld + k0 + col;
        r.lo = *(const int4*)(p + 0);
        r.hi = *(const int4*)(p + 8);
    }
    return r;
}

__device__ inline void store_stage_seg(_Float16* dst, int seg, StageRegs r) {
    const int row = seg >> 1;
    const int col = (seg & 1) * 16;
    *(int4*)(dst + row * LDH + col + 0) = r.lo;
    *(int4*)(dst + row * LDH + col + 8) = r.hi;
}

// Async global->LDS staging of a ROWSx32-half tile, 16B per lane-issue.
// Tracked by ASYNCcnt; pair with s_wait_asynccnt 0 + barrier before compute.
template<int ROWS>
__device__ inline void async_stage(const _Float16* src, int ld, size_t rowOff,
                                   int k0, _Float16* dst, int tid) {
    #pragma unroll
    for (int i = 0; i < (ROWS * 4) / 256; ++i) {
        const int s   = tid + i * 256;
        const int row = s >> 2;
        const int cs  = s & 3;                    // 8-half (16B) column segment
        const unsigned ldsa = (unsigned)(size_t)(dst + row * LDH + cs * 8);
        const _Float16* g = src + (rowOff + (size_t)row) * (size_t)ld + k0 + cs * 8;
        asm volatile("global_load_async_to_lds_b128 %0, %1, off"
                     :: "v"(ldsa), "v"((unsigned long long)(size_t)g)
                     : "memory");
    }
}

__device__ inline void wait_async0() {
    asm volatile("s_wait_asynccnt 0" ::: "memory");
}

// 16-bit A/B operand fragment per the CDNA5 ISA layout:
// lanes 0-15 hold row (m&15), K 0..7 and 16..23; lanes 16-31 hold K 8..15 and 24..31.
__device__ inline v16h load_frag(const _Float16* ldsRow, int laneHi) {
    union { v16h v; int4 q[2]; } u;
    u.q[0] = *(const int4*)(ldsRow + laneHi * 8);
    u.q[1] = *(const int4*)(ldsRow + 16 + laneHi * 8);
    return u.v;
}

__device__ inline v8f zero8() {
    v8f v;
    #pragma unroll
    for (int i = 0; i < 8; ++i) v[i] = 0.0f;
    return v;
}

template<int MODE>
__global__ __launch_bounds__(256) void fa_gemm_wmma(
    const void* __restrict__ Aall, const void* __restrict__ Ball,
    const float* __restrict__ bias, void* __restrict__ Cout,
    int Kd, int lda, int ldb, int ldc,
    long long strideA, long long strideB, long long strideC, float scale) {

    constexpr bool AF32 = (MODE == MODE_PROJ || MODE == MODE_PROJV);

    __shared__ alignas(16) _Float16 sA[2][BM * LDH];   // 40 KB
    __shared__ alignas(16) _Float16 sB[2][BN * LDH];   // 20 KB

    const int tid     = threadIdx.x;
    const int wave    = tid >> 5;
    const int lane    = tid & 31;
    const int laneRow = lane & 15;
    const int laneHi  = lane >> 4;         // 0/1 -> K-half selector & M+8 selector
    const int wm0     = (wave & 3) * 64;   // 4 waves along M, 64 rows each
    const int wn0     = (wave >> 2) * 64;  // 2 waves along N, 64 cols each

    const size_t row0 = (size_t)blockIdx.y * BM;   // within batch z
    const size_t col0 = (size_t)blockIdx.x * BN;

    const void* Ab;
    const void* Bb;
    if constexpr (AF32) {
        Ab = (const void*)((const float*)Aall + (size_t)blockIdx.z * strideA);
        Bb = (const void*)((const float*)Ball + (size_t)blockIdx.z * strideB);
    } else {
        Ab = (const void*)((const _Float16*)Aall + (size_t)blockIdx.z * strideA);
        Bb = (const void*)((const _Float16*)Ball + (size_t)blockIdx.z * strideB);
    }

    v8f acc[4][4];
    #pragma unroll
    for (int mt = 0; mt < 4; ++mt)
        #pragma unroll
        for (int nt = 0; nt < 4; ++nt) acc[mt][nt] = zero8();

    const int KT = Kd / BK;

    auto compute = [&](int buf) {
        v16h bfrag[4];
        #pragma unroll
        for (int nt = 0; nt < 4; ++nt)
            bfrag[nt] = load_frag(&sB[buf][(wn0 + nt * 16 + laneRow) * LDH], laneHi);
        #pragma unroll
        for (int mt = 0; mt < 4; ++mt) {
            const v16h af = load_frag(&sA[buf][(wm0 + mt * 16 + laneRow) * LDH], laneHi);
            #pragma unroll
            for (int nt = 0; nt < 4; ++nt)
                acc[mt][nt] = __builtin_amdgcn_wmma_f32_16x16x32_f16(
                    false, af, false, bfrag[nt], (short)0, acc[mt][nt], false, false);
        }
    };

    if constexpr (AF32) {
        // f32 sources: register staging with f32->f16 convert, double buffered.
        StageRegs ra0 = load_stage_seg<true>(Ab, lda, row0, 0, tid);
        StageRegs ra1 = load_stage_seg<true>(Ab, lda, row0, 0, tid + 256);
        StageRegs rb0 = load_stage_seg<true>(Bb, ldb, col0, 0, tid);
        int buf = 0;
        for (int kt = 0; kt < KT; ++kt) {
            store_stage_seg(sA[buf], tid,       ra0);
            store_stage_seg(sA[buf], tid + 256, ra1);
            store_stage_seg(sB[buf], tid,       rb0);
            __syncthreads();
            if (kt + 1 < KT) {   // prefetch next tile under the WMMA block
                ra0 = load_stage_seg<true>(Ab, lda, row0, (kt + 1) * BK, tid);
                ra1 = load_stage_seg<true>(Ab, lda, row0, (kt + 1) * BK, tid + 256);
                rb0 = load_stage_seg<true>(Bb, ldb, col0, (kt + 1) * BK, tid);
            }
            compute(buf);
            buf ^= 1;
        }
    } else {
        // f16 sources: async global->LDS DMA, double buffered on ASYNCcnt.
        const _Float16* Ah = (const _Float16*)Ab;
        const _Float16* Bh = (const _Float16*)Bb;
        async_stage<BM>(Ah, lda, row0, 0, sA[0], tid);
        async_stage<BN>(Bh, ldb, col0, 0, sB[0], tid);
        int buf = 0;
        for (int kt = 0; kt < KT; ++kt) {
            wait_async0();
            __syncthreads();
            if (kt + 1 < KT) {   // async loads for kt+1 land under the WMMAs
                async_stage<BM>(Ah, lda, row0, (kt + 1) * BK, sA[buf ^ 1], tid);
                async_stage<BN>(Bh, ldb, col0, (kt + 1) * BK, sB[buf ^ 1], tid);
            }
            compute(buf);
            buf ^= 1;
        }
    }

    // Epilogue. C tile layout: N = lane&15, M = r + 8*(lane>>4) per VGPR r.
    #pragma unroll
    for (int mt = 0; mt < 4; ++mt) {
        #pragma unroll
        for (int nt = 0; nt < 4; ++nt) {
            const size_t n  = col0 + wn0 + nt * 16 + laneRow;
            const size_t mb = row0 + wm0 + mt * 16 + laneHi * 8;
            float badd = 0.0f;
            if constexpr (MODE == MODE_PROJ || MODE == MODE_PROJV) badd = bias[n];

            if constexpr (MODE == MODE_PROJ) {
                _Float16* cp = (_Float16*)Cout + mb * (size_t)ldc + n;
                #pragma unroll
                for (int r = 0; r < 8; ++r)
                    cp[(size_t)r * ldc] = (_Float16)(acc[mt][nt][r] + badd);
            } else if constexpr (MODE == MODE_PROJV) {
                // Write V transposed: Vt[b][u][s], 8 contiguous f16 -> one b128 store.
                union { int4 q; _Float16 h[8]; } u;
                #pragma unroll
                for (int r = 0; r < 8; ++r)
                    u.h[r] = (_Float16)(acc[mt][nt][r] + badd);
                const size_t b = mb >> 11;           // mb / 2048
                const size_t s = mb & 2047;
                *(int4*)((_Float16*)Cout + ((b << 10) + n) * (size_t)Ssz + s) = u.q;
            } else if constexpr (MODE == MODE_SCORES) {
                _Float16* cp = (_Float16*)Cout + (size_t)blockIdx.z * strideC
                               + mb * (size_t)ldc + n;
                #pragma unroll
                for (int r = 0; r < 8; ++r)
                    cp[(size_t)r * ldc] = (_Float16)(acc[mt][nt][r] * scale);
            } else { // MODE_AV -> f32 final output
                float* cp = (float*)Cout + (size_t)blockIdx.z * strideC
                            + mb * (size_t)ldc + n;
                #pragma unroll
                for (int r = 0; r < 8; ++r)
                    cp[(size_t)r * ldc] = acc[mt][nt][r];
            }
        }
    }
}

// Row softmax in-place on P (f16), one 256-thread block per 2048-wide row.
__global__ __launch_bounds__(256) void fa_softmax(_Float16* __restrict__ P) {
    __shared__ float red[256];
    _Float16* p = P + (size_t)blockIdx.x * Ssz;
    const int t = threadIdx.x;

    float v[8];
    #pragma unroll
    for (int i = 0; i < 8; ++i) v[i] = (float)p[t + i * 256];

    float mx = v[0];
    #pragma unroll
    for (int i = 1; i < 8; ++i) mx = fmaxf(mx, v[i]);
    red[t] = mx; __syncthreads();
    for (int s = 128; s > 0; s >>= 1) {
        if (t < s) red[t] = fmaxf(red[t], red[t + s]);
        __syncthreads();
    }
    mx = red[0]; __syncthreads();

    float sum = 0.0f;
    #pragma unroll
    for (int i = 0; i < 8; ++i) { v[i] = __expf(v[i] - mx); sum += v[i]; }
    red[t] = sum; __syncthreads();
    for (int s = 128; s > 0; s >>= 1) {
        if (t < s) red[t] += red[t + s];
        __syncthreads();
    }
    const float inv = 1.0f / red[0];
    #pragma unroll
    for (int i = 0; i < 8; ++i) p[t + i * 256] = (_Float16)(v[i] * inv);
}

extern "C" void kernel_launch(void* const* d_in, const int* in_sizes, int n_in,
                              void* d_out, int out_size, void* d_ws, size_t ws_size,
                              hipStream_t stream) {
    (void)in_sizes; (void)n_in; (void)out_size; (void)ws_size;
    const float* x  = (const float*)d_in[0];
    const float* Wq = (const float*)d_in[1];
    const float* bq = (const float*)d_in[2];
    const float* Wk = (const float*)d_in[3];
    const float* bk = (const float*)d_in[4];
    const float* Wv = (const float*)d_in[5];
    const float* bv = (const float*)d_in[6];
    float* out = (float*)d_out;

    const size_t M  = (size_t)Bsz * Ssz;       // 8192
    _Float16* Qh = (_Float16*)d_ws;            // [B*S, U]        16 MB
    _Float16* Kh = Qh + M * Usz;               // [B*S, U]        16 MB
    _Float16* Vt = Kh + M * Usz;               // [B, U, S]       16 MB
    _Float16* P  = Vt + M * Usz;               // [B, Sq, Sk]     32 MB

    const float sscale = 1.0f / 32.0f;         // 1/sqrt(1024)
    dim3 blk(256);

    // Q/K/V projections: M=8192, N=1024, K=1024 (f32 in, f16 out)
    dim3 gp(Usz / BN, M / BM, 1);
    fa_gemm_wmma<MODE_PROJ ><<<gp, blk, 0, stream>>>(x, Wq, bq, Qh, Dsz, Dsz, Dsz, Usz, 0, 0, 0, 1.0f);
    fa_gemm_wmma<MODE_PROJ ><<<gp, blk, 0, stream>>>(x, Wk, bk, Kh, Dsz, Dsz, Dsz, Usz, 0, 0, 0, 1.0f);
    fa_gemm_wmma<MODE_PROJV><<<gp, blk, 0, stream>>>(x, Wv, bv, Vt, Dsz, Dsz, Dsz, Usz, 0, 0, 0, 1.0f);

    // scores = Q K^T / 32 per batch: M=N=2048, K=1024 (async-LDS staging)
    dim3 gs(Ssz / BN, Ssz / BM, Bsz);
    fa_gemm_wmma<MODE_SCORES><<<gs, blk, 0, stream>>>(
        Qh, Kh, nullptr, P, Usz, Usz, Usz, Ssz,
        (long long)Ssz * Usz, (long long)Ssz * Usz, (long long)Ssz * Ssz, sscale);

    // softmax over each of the B*S rows of P
    fa_softmax<<<dim3((unsigned)M), blk, 0, stream>>>(P);

    // out = P V per batch (via V^T rows): M=2048, N=1024, K=2048 (async staging)
    dim3 ga(Usz / BN, Ssz / BM, Bsz);
    fa_gemm_wmma<MODE_AV><<<ga, blk, 0, stream>>>(
        P, Vt, nullptr, out, Ssz, Ssz, Ssz, Usz,
        (long long)Ssz * Ssz, (long long)Usz * Ssz, (long long)Ssz * Usz, 1.0f);
}
